// AlignBlock_12506944766241
// MI455X (gfx1250) — matrix operand order repaired and checked by reading, not compiled, not run on hardware
//
#include <hip/hip_runtime.h>
#include <hip/hip_bf16.h>
#include <stdint.h>
#include <stddef.h>

typedef __attribute__((ext_vector_type(16))) __bf16 v16bf;
typedef __attribute__((ext_vector_type(8)))  float  v8f;

#define BB 2
#define CC 128
#define TT 512
#define FF 64
#define HH 32
#define DD 32
#define NN (TT * FF) // 32768 columns per (b, input) in the projection GEMM

__device__ __forceinline__ v8f wmma_bf16(v16bf a, v16bf b, v8f c) {
  // D = A(16x32 bf16) * B(32x16 bf16) + C(16x16 f32)
  return __builtin_amdgcn_wmma_f32_16x16x32_bf16(
      /*neg_a=*/false, a, /*neg_b=*/false, b,
      /*c_mod=*/(short)0, c, /*reuse_a=*/false, /*reuse_b=*/false);
}

// ---------------------------------------------------------------------------
// Stage 1: y[b][h][n] = sum_c w[c][h] * x[b][c][n] + bias[h], n = t*64 + f
// GEMM 32 x 128 x 32768 per (b, input). Output stored as bf16 [b][h][t][f].
// One wave owns an 8-tile n-strip; A (w^T, all 4 K-chunks, both h-tiles)
// lives in registers across the strip.
// ---------------------------------------------------------------------------
__global__ __launch_bounds__(128) void proj_kernel(
    const float* __restrict__ xm, const float* __restrict__ xf,
    const float* __restrict__ w1, const float* __restrict__ b1,
    const float* __restrict__ w2, const float* __restrict__ b2,
    __bf16* __restrict__ xm1, __bf16* __restrict__ xf1) {
  const int wave = blockIdx.x * 4 + (threadIdx.x >> 5);
  const int lane = threadIdx.x & 31;
  const int half = lane >> 4;
  const int mn   = lane & 15;

  const int strips = (NN / 16) / 8;           // 256 strips per (b, input)
  const int strip  = wave % strips;
  const int input  = (wave / strips) & 1;
  const int b      = wave / (strips * 2);

  const float* x    = input ? xf : xm;
  const float* w    = input ? w2 : w1;
  const float* bias = input ? b2 : b1;
  __bf16*      y    = input ? xf1 : xm1;

  // A[m][k] = w[k][hbase+m]  (16-bit A layout: lane m, VGPR v -> K pair)
  v16bf a[2][4];
#pragma unroll
  for (int ht = 0; ht < 2; ++ht) {
#pragma unroll
    for (int kc = 0; kc < 4; ++kc) {
#pragma unroll
      for (int v = 0; v < 8; ++v) {
        const int kb   = kc * 32 + ((v >= 4) ? 16 : 0) + half * 8 + (v & 3) * 2;
        const int hcol = ht * 16 + mn;
        a[ht][kc][2 * v]     = (__bf16)w[(size_t)(kb)     * HH + hcol];
        a[ht][kc][2 * v + 1] = (__bf16)w[(size_t)(kb + 1) * HH + hcol];
      }
    }
  }

  const float* xb = x + (size_t)b * CC * NN;
  __bf16*      yb = y + (size_t)b * HH * NN;

  for (int s = 0; s < 8; ++s) {
    const int nbase = (strip * 8 + s) * 16;
    v8f acc0 = {0.f, 0.f, 0.f, 0.f, 0.f, 0.f, 0.f, 0.f};
    v8f acc1 = {0.f, 0.f, 0.f, 0.f, 0.f, 0.f, 0.f, 0.f};
#pragma unroll
    for (int kc = 0; kc < 4; ++kc) {
      v16bf bb; // B[k][n'] = x[c = kc*32+k][nbase+n']
#pragma unroll
      for (int v = 0; v < 8; ++v) {
        const int kb = kc * 32 + ((v >= 4) ? 16 : 0) + half * 8 + (v & 3) * 2;
        bb[2 * v]     = (__bf16)xb[(size_t)(kb)     * NN + nbase + mn];
        bb[2 * v + 1] = (__bf16)xb[(size_t)(kb + 1) * NN + nbase + mn];
      }
      acc0 = wmma_bf16(a[0][kc], bb, acc0);
      acc1 = wmma_bf16(a[1][kc], bb, acc1);
    }
#pragma unroll
    for (int r = 0; r < 8; ++r) {
      const int m = r + half * 8; // C/D layout: VGPR r -> row r (lanes 0-15) / r+8
      yb[(size_t)(m)      * NN + nbase + mn] = (__bf16)(acc0[r] + bias[m]);
      yb[(size_t)(16 + m) * NN + nbase + mn] = (__bf16)(acc1[r] + bias[16 + m]);
    }
  }
}

// ---------------------------------------------------------------------------
// Stage 2: corr[b][h][t][d] = sum_f xm1[b][h][t][f] * xf1[b][h][t-31+d][f]
// Band of the t x t Gram matrix: per 16-row t-block, 3 column tiles cover
// times Tb-32 .. Tb+15.  K = f = 64 (2 WMMA chunks).  t<0 columns are zero.
// ---------------------------------------------------------------------------
__global__ __launch_bounds__(128) void corr_kernel(
    const __bf16* __restrict__ xm1, const __bf16* __restrict__ xf1,
    float* __restrict__ corr) {
  const int wave = blockIdx.x * 4 + (threadIdx.x >> 5);
  const int lane = threadIdx.x & 31;
  const int half = lane >> 4;
  const int mn   = lane & 15;

  const int tt = wave & 31;
  const int h  = (wave >> 5) & 31;
  const int b  = wave >> 10;
  const int Tb = tt * 16;

  const __bf16* am = xm1 + (size_t)(b * HH + h) * TT * FF;
  const __bf16* bm = xf1 + (size_t)(b * HH + h) * TT * FF;
  float*        cr = corr + (size_t)(b * HH + h) * TT * DD;

  v16bf a[2]; // A[m][f] = xm1[Tb+m][f]
#pragma unroll
  for (int kc = 0; kc < 2; ++kc) {
#pragma unroll
    for (int v = 0; v < 8; ++v) {
      const int fb = kc * 32 + ((v >= 4) ? 16 : 0) + half * 8 + (v & 3) * 2;
      a[kc][2 * v]     = am[(size_t)(Tb + mn) * FF + fb];
      a[kc][2 * v + 1] = am[(size_t)(Tb + mn) * FF + fb + 1];
    }
  }

  const __bf16 zero = (__bf16)0.0f;
#pragma unroll
  for (int ctile = 0; ctile < 3; ++ctile) {
    const int Tc   = Tb - 32 + ctile * 16;
    const int time = Tc + mn; // this lane's B column time index
    v8f acc = {0.f, 0.f, 0.f, 0.f, 0.f, 0.f, 0.f, 0.f};
#pragma unroll
    for (int kc = 0; kc < 2; ++kc) {
      v16bf bb; // B[f][j] = xf1[Tc+j][f]
#pragma unroll
      for (int v = 0; v < 8; ++v) {
        const int fb = kc * 32 + ((v >= 4) ? 16 : 0) + half * 8 + (v & 3) * 2;
        bb[2 * v]     = (time >= 0) ? bm[(size_t)time * FF + fb]     : zero;
        bb[2 * v + 1] = (time >= 0) ? bm[(size_t)time * FF + fb + 1] : zero;
      }
      acc = wmma_bf16(a[kc], bb, acc);
    }
#pragma unroll
    for (int r = 0; r < 8; ++r) {
      const int m = r + half * 8;
      const int d = ctile * 16 + mn - m - 1; // d = (Tc+j) - (Tb+m) + 31
      if ((unsigned)d < 32u)
        cr[(size_t)(Tb + m) * DD + d] = acc[r];
    }
  }
}

// ---------------------------------------------------------------------------
// Stage 3: 5x3 conv over (h, t, d) -> 1 channel, +bc, causal mask, softmax(d).
// One wave per (b, t); lane = d.
// ---------------------------------------------------------------------------
__global__ __launch_bounds__(256) void softmax_kernel(
    const float* __restrict__ corr, const float* __restrict__ wc,
    const float* __restrict__ bc, __bf16* __restrict__ wbuf) {
  const int wave = blockIdx.x * 8 + (threadIdx.x >> 5);
  const int d    = threadIdx.x & 31;
  const int t    = wave & (TT - 1);
  const int b    = wave >> 9;

  float s = bc[0];
  for (int h = 0; h < HH; ++h) {
    const float* crh = corr + (size_t)(b * HH + h) * TT * DD;
#pragma unroll
    for (int i = 0; i < 5; ++i) {
      const int t2 = t + i - 3;
      if (t2 < 0 || t2 >= TT) continue;
      const float* row = crh + (size_t)t2 * DD;
#pragma unroll
      for (int j = 0; j < 3; ++j) {
        const int dd = d + j - 1;
        if ((unsigned)dd < 32u) s += row[dd] * wc[h * 15 + i * 3 + j];
      }
    }
  }
  if (t + d < DD - 1) s = -1.0e13f; // NEG_INF mask

  float mx = s;
#pragma unroll
  for (int off = 16; off >= 1; off >>= 1) mx = fmaxf(mx, __shfl_xor(mx, off, 32));
  const float e = __expf(s - mx);
  float sum = e;
#pragma unroll
  for (int off = 16; off >= 1; off >>= 1) sum += __shfl_xor(sum, off, 32);

  wbuf[(size_t)(b * TT + t) * DD + d] = (__bf16)(e / sum);
}

// ---------------------------------------------------------------------------
// Stage 4: out[b][c][t][f] = sum_d w[b][t][d] * xf[b][c][t-31+d][f]
// Banded (t x t) x (t x f) matmul: A split into strictly-upper chunk
// (times Tb-32..Tb-1) and lower-incl-diag chunk (times Tb..Tb+31).
// One wave per (b, c, 16-t-block); A reused across 4 f-tiles.
// ---------------------------------------------------------------------------
__global__ __launch_bounds__(128) void out_kernel(
    const float* __restrict__ xf, const __bf16* __restrict__ wbuf,
    float* __restrict__ out) {
  const int wave = blockIdx.x * 4 + (threadIdx.x >> 5);
  const int lane = threadIdx.x & 31;
  const int half = lane >> 4;
  const int mn   = lane & 15;

  const int tt = wave & 31;
  const int c  = (wave >> 5) & 127;
  const int b  = wave >> 12;
  const int Tb = tt * 16;

  const __bf16* wrow = wbuf + (size_t)(b * TT + Tb + mn) * DD; // A row m = mn
  const __bf16 zero = (__bf16)0.0f;

  v16bf a1, a2;
#pragma unroll
  for (int v = 0; v < 8; ++v) {
    const int kb = ((v >= 4) ? 16 : 0) + half * 8 + (v & 3) * 2;
#pragma unroll
    for (int q = 0; q < 2; ++q) {
      const int k = kb + q;
      // chunk 1: time = Tb-32+k -> d = k-m-1, valid when k >= m+1
      a1[2 * v + q] = (k >= mn + 1) ? wrow[k - mn - 1] : zero;
      // chunk 2: time = Tb+k    -> d = k-m+31, valid when k <= m
      a2[2 * v + q] = (k <= mn) ? wrow[k - mn + 31] : zero;
    }
  }

  const float* xc = xf + (size_t)(b * CC + c) * TT * FF;
  float*       oc = out + (size_t)(b * CC + c) * TT * FF;

#pragma unroll
  for (int ft = 0; ft < 4; ++ft) {
    v8f acc = {0.f, 0.f, 0.f, 0.f, 0.f, 0.f, 0.f, 0.f};
#pragma unroll
    for (int cc = 0; cc < 2; ++cc) {
      v16bf bb; // B[k][n] = xf[time][ft*16+n]
#pragma unroll
      for (int v = 0; v < 8; ++v) {
        const int k  = ((v >= 4) ? 16 : 0) + half * 8 + (v & 3) * 2;
        const int t0 = Tb - 32 + cc * 32 + k;
        const int t1 = t0 + 1;
        bb[2 * v]     = ((unsigned)t0 < (unsigned)TT)
                            ? (__bf16)xc[(size_t)t0 * FF + ft * 16 + mn] : zero;
        bb[2 * v + 1] = ((unsigned)t1 < (unsigned)TT)
                            ? (__bf16)xc[(size_t)t1 * FF + ft * 16 + mn] : zero;
      }
      acc = wmma_bf16(cc == 0 ? a1 : a2, bb, acc);
    }
#pragma unroll
    for (int r = 0; r < 8; ++r) {
      const int m = r + half * 8;
      oc[(size_t)(Tb + m) * FF + ft * 16 + mn] = acc[r];
    }
  }
}

// ---------------------------------------------------------------------------
extern "C" void kernel_launch(void* const* d_in, const int* in_sizes, int n_in,
                              void* d_out, int out_size, void* d_ws,
                              size_t ws_size, hipStream_t stream) {
  (void)in_sizes; (void)n_in; (void)out_size; (void)ws_size;
  const float* xm = (const float*)d_in[0];
  const float* xf = (const float*)d_in[1];
  const float* w1 = (const float*)d_in[2];
  const float* b1 = (const float*)d_in[3];
  const float* w2 = (const float*)d_in[4];
  const float* b2 = (const float*)d_in[5];
  const float* wc = (const float*)d_in[6];
  const float* bc = (const float*)d_in[7];

  char* ws = (char*)d_ws;
  __bf16* xm1  = (__bf16*)(ws);                          // 4 MB bf16 [b,h,t,f]
  __bf16* xf1  = (__bf16*)(ws + ((size_t)4 << 20));      // 4 MB bf16 [b,h,t,f]
  float*  corr = (float*) (ws + ((size_t)8 << 20));      // 4 MB f32  [b,h,t,d]
  __bf16* wbuf = (__bf16*)(ws + ((size_t)12 << 20));     // 64 KB bf16 [b,t,d]

  // 1024 wave-tasks = 2 b * 2 inputs * 256 strips
  proj_kernel<<<256, 128, 0, stream>>>(xm, xf, w1, b1, w2, b2, xm1, xf1);
  // 2048 wave-tasks = 2 b * 32 h * 32 t-tiles
  corr_kernel<<<512, 128, 0, stream>>>(xm1, xf1, corr);
  // 1024 wave-tasks = 2 b * 512 t
  softmax_kernel<<<128, 256, 0, stream>>>(corr, wc, bc, wbuf);
  // 8192 wave-tasks = 2 b * 128 c * 32 t-tiles
  out_kernel<<<2048, 128, 0, stream>>>(xf, wbuf, (float*)d_out);
}
